// MultiHeadAttention_49392123904572
// MI455X (gfx1250) — matrix operand order repaired and checked by reading, not compiled, not run on hardware
//
#include <hip/hip_runtime.h>
#include <hip/hip_bf16.h>

// ---------------------------------------------------------------------------
// MultiHeadAttention forward for gfx1250 (MI455X), wave32 + WMMA bf16.
// B=2, S=2048, D=1024, H=16, HD=64.
// d_out: x (B,S,D) f32  ||  attn (B,H,S,S) f32.
// Pipeline: 3x projection GEMM (WMMA, double-buffered async-LDS staging) ->
//           fused logits+mask+softmax+ctx (attn written to HBM exactly once) ->
//           output projection GEMM.
// ---------------------------------------------------------------------------

#define BB 2
#define SS 2048
#define DD 1024
#define HH 16
#define HD 64

typedef __attribute__((ext_vector_type(16))) __bf16 v16bf;
typedef __attribute__((ext_vector_type(8)))  __bf16 v8bf;
typedef __attribute__((ext_vector_type(8)))  float  v8f;
typedef __attribute__((ext_vector_type(4)))  float  v4f;
typedef __attribute__((ext_vector_type(4)))  int    v4i;

__device__ __forceinline__ v8bf cvt8(v8f x) {
    v8bf r;
#pragma unroll
    for (int i = 0; i < 8; ++i) r[i] = (__bf16)x[i];
    return r;
}

// A-fragment gather: 16 contiguous elements relative to base
__device__ __forceinline__ v16bf fragA(const float* base) {
    union { v16bf v; v8bf h[2]; } u;
    u.h[0] = cvt8(*(const v8f*)base);
    u.h[1] = cvt8(*(const v8f*)(base + 8));
    return u.v;
}
__device__ __forceinline__ v16bf fragA(const __bf16* base) {
    union { v16bf v; v8bf h[2]; } u;
    u.h[0] = *(const v8bf*)base;
    u.h[1] = *(const v8bf*)(base + 8);
    return u.v;
}
// B-fragment gather: 16 contiguous elements
__device__ __forceinline__ v16bf fragB(const float* base) {
    union { v16bf v; v8bf h[2]; } u;
    u.h[0] = cvt8(*(const v8f*)base);
    u.h[1] = cvt8(*(const v8f*)(base + 8));
    return u.v;
}

__device__ __forceinline__ v8f wmma_bf16(v16bf a, v16bf b, v8f c) {
    return __builtin_amdgcn_wmma_f32_16x16x32_bf16(
        false, a, false, b, (short)0, c, false, false);
}

// 16-byte async copy global -> LDS (ASYNCcnt-tracked, §10 async LDS ops)
__device__ __forceinline__ void async_copy16(void* lds, const void* g) {
    unsigned l = (unsigned)(unsigned long long)lds;   // flat LDS addr[31:0] = LDS offset
    asm volatile("global_load_async_to_lds_b128 %0, %1, off"
                 :: "v"(l), "v"((unsigned long long)g) : "memory");
}
__device__ __forceinline__ void async_wait0() {
    asm volatile("s_wait_asynccnt 0x0" ::: "memory");
}

__device__ __forceinline__ float wredmax(float x) {
#pragma unroll
    for (int o = 16; o > 0; o >>= 1) x = fmaxf(x, __shfl_xor(x, o, 32));
    return x;
}
__device__ __forceinline__ float wredsum(float x) {
#pragma unroll
    for (int o = 16; o > 0; o >>= 1) x += __shfl_xor(x, o, 32);
    return x;
}

// ---------------------------------------------------------------------------
// GEMM: C[m,n] = sum_k A[m,k]*W[n,k] + bias[n];  K=N=1024.
// MODE 0: bf16 out (b,h,s,d) | MODE 1: bf16 out (b,h,d,s) | MODE 2: f32 out (m,n)
// Block 256 thr (8 waves), block tile 128x128, wave tile 16x128, K-step 32.
// Double-buffered LDS: tile t+1 streams in via global_load_async_to_lds_b128
// while tile t's WMMAs execute; s_wait_asynccnt 0 + barrier closes the stage.
// ---------------------------------------------------------------------------
template <typename AT, int MODE>
__global__ void gemm_wmma(const AT* __restrict__ A, const float* __restrict__ W,
                          const float* __restrict__ bias, void* __restrict__ outp) {
    constexpr int ROWPA = (sizeof(AT) == 4) ? 36 : 40;   // padded row pitch (elems)
    constexpr int CPRA  = (32 * sizeof(AT)) / 16;        // 16B chunks per tile row
    constexpr int EPCA  = 16 / sizeof(AT);               // elems per 16B chunk
    constexpr int NT    = 1024 / 32;                     // K tiles
    __shared__ __align__(64) AT    lsA[2][128 * ROWPA];
    __shared__ __align__(64) float lsB[2][128 * 36];

    const int tid = threadIdx.x;
    const int wv  = tid >> 5;
    const int ln  = tid & 31;
    const int l15 = ln & 15;
    const int lh  = ln >> 4;
    const int nBase = blockIdx.x * 128;
    const int mBase = blockIdx.y * 128;

    auto stage = [&](int k0, int buf) {
#pragma unroll
        for (int i = 0; i < (128 * CPRA) / 256; ++i) {
            const int chunk = tid + i * 256;
            const int row   = chunk / CPRA;
            const int ce    = (chunk % CPRA) * EPCA;
            async_copy16(&lsA[buf][row * ROWPA + ce],
                         &A[(size_t)(mBase + row) * 1024 + k0 + ce]);
        }
#pragma unroll
        for (int i = 0; i < 4; ++i) {
            const int chunk = tid + i * 256;
            const int row   = chunk >> 3;
            const int ce    = (chunk & 7) * 4;
            async_copy16(&lsB[buf][row * 36 + ce],
                         &W[(size_t)(nBase + row) * 1024 + k0 + ce]);
        }
    };

    v8f acc[8] = {};

    stage(0, 0);
    async_wait0();
    __syncthreads();

    for (int t = 0; t < NT; ++t) {
        const int cur = t & 1;
        // stream next tile into the other buffer while computing this one
        if (t + 1 < NT) stage((t + 1) * 32, cur ^ 1);
        if (t + 2 < NT) {   // keep L2 primed two tiles ahead (global_prefetch_b8)
            __builtin_prefetch(&A[(size_t)(mBase + (tid & 127)) * 1024 + (t + 2) * 32], 0, 1);
            __builtin_prefetch(&W[(size_t)(nBase + (tid & 127)) * 1024 + (t + 2) * 32], 0, 1);
        }

        // A fragment 16x32: lanes<16 K0..15; lanes>=16 K16..31
        const v16bf va = fragA(&lsA[cur][(wv * 16 + l15) * ROWPA + lh * 16]);
#pragma unroll
        for (int c = 0; c < 8; ++c) {
            // B fragment 32x16: lane n = col n; lanes<16 K0..15, lanes>=16 K16..31
            const v16bf vb = fragB(&lsB[cur][(c * 16 + l15) * 36 + lh * 16]);
            acc[c] = wmma_bf16(va, vb, acc[c]);
        }

        // next-tile copies complete + all waves done reading before buffer swap
        async_wait0();
        __syncthreads();
    }

    // D layout: lane l, vgpr r -> (M = r + 8*(l>=16), N = l&15)
#pragma unroll
    for (int c = 0; c < 8; ++c) {
        const int   n  = nBase + c * 16 + l15;
        const float bv = bias[n];
#pragma unroll
        for (int r = 0; r < 8; ++r) {
            const int   m   = mBase + wv * 16 + r + lh * 8;
            const float val = acc[c][r] + bv;
            if (MODE == 0) {
                __bf16* out = (__bf16*)outp;
                const int b = m >> 11, s = m & (SS - 1);
                const int h = n >> 6,  d = n & (HD - 1);
                out[((size_t)(b * HH + h) * SS + s) * HD + d] = (__bf16)val;
            } else if (MODE == 1) {
                __bf16* out = (__bf16*)outp;
                const int b = m >> 11, s = m & (SS - 1);
                const int h = n >> 6,  d = n & (HD - 1);
                out[((size_t)(b * HH + h) * HD + d) * SS + s] = (__bf16)val;
            } else {
                float* out = (float*)outp;
                out[(size_t)m * DD + n] = val;
            }
        }
    }
}

// ---------------------------------------------------------------------------
// Fused attention core. One block = 16 q-rows x full S, one (b,h).
// Phase 1: logits (WMMA, Q/K bf16) -> LDS slab (16 x 2048 f32, padded pitch)
// Phase 2: mask + row softmax in LDS (coalesced int4 mask loads),
//          final attn written to d_out once (never re-read from HBM).
// Phase 3: ctx = attn @ V (WMMA, A from LDS f32->bf16, V transposed layout),
//          K split across 8 waves, partials tree-reduced in LDS.
// LDS: 16*2052*4 + 8*16*64*4 = 164096 B (dynamic) -> 1 block / WGP (320KB).
// K/V per head = 512KB, whole K/V set = 16MB -> re-reads are 192MB-L2 hits.
// ---------------------------------------------------------------------------
#define LROW 2052

__global__ void attn_fused(const __bf16* __restrict__ Qh,
                           const __bf16* __restrict__ Kh,
                           const __bf16* __restrict__ Vt,
                           const int* __restrict__ mask,
                           float* __restrict__ attnOut,
                           __bf16* __restrict__ ctxOut) {
    extern __shared__ char smem[];
    float* lgt  = (float*)smem;                                  // 16 x LROW
    float* rbuf = (float*)(smem + 16 * LROW * sizeof(float));    // 8 x 16 x 64

    const int tid = threadIdx.x;
    const int wv  = tid >> 5;
    const int ln  = tid & 31;
    const int l15 = ln & 15;
    const int lh  = ln >> 4;
    const int bh  = blockIdx.y;
    const int b   = bh >> 4;
    const int h   = bh & 15;
    const int qBase = blockIdx.x * 16;

    const __bf16* Qb = Qh + (size_t)bh * SS * HD;
    const __bf16* Kb = Kh + (size_t)bh * SS * HD;
    const __bf16* Vb = Vt + (size_t)bh * HD * SS;

    // ---- Phase 1: logits for cols [wv*256, wv*256+256) ----
    union { v16bf v; v8bf p[2]; } a0, a1;
    {
        const __bf16* qrow = &Qb[(size_t)(qBase + l15) * HD];
        a0.p[0] = *(const v8bf*)&qrow[lh * 16];
        a0.p[1] = *(const v8bf*)&qrow[lh * 16 + 8];
        a1.p[0] = *(const v8bf*)&qrow[32 + lh * 16];
        a1.p[1] = *(const v8bf*)&qrow[32 + lh * 16 + 8];
    }
#pragma unroll
    for (int ct = 0; ct < 16; ++ct) {
        const int n0 = wv * 256 + ct * 16;
        const __bf16* krow = &Kb[(size_t)(n0 + l15) * HD];
        const v16bf b0 = *(const v16bf*)&krow[lh * 16];
        const v16bf b1 = *(const v16bf*)&krow[32 + lh * 16];
        v8f acc = {};
        acc = wmma_bf16(a0.v, b0, acc);
        acc = wmma_bf16(a1.v, b1, acc);
        const int n = n0 + l15;
#pragma unroll
        for (int r = 0; r < 8; ++r)
            lgt[(r + lh * 8) * LROW + n] = acc[r] * 0.125f;
    }
    __syncthreads();

    // ---- Phase 2: mask + softmax, 2 rows per wave ----
#pragma unroll
    for (int rr = 0; rr < 2; ++rr) {
        const int row = wv * 2 + rr;
        const int q   = qBase + row;
        float*      Lr = &lgt[row * LROW];
        const int*  Mr = &mask[(size_t)(b * SS + q) * SS];

        float mx = -3.4e38f;
#pragma unroll 4
        for (int it = 0; it < 16; ++it) {
            const int c = it * 128 + ln * 4;
            v4f x  = *(v4f*)&Lr[c];
            v4i mk = *(const v4i*)&Mr[c];
#pragma unroll
            for (int j = 0; j < 4; ++j) {
                x[j] += (float)mk[j] * -1e9f;
                mx = fmaxf(mx, x[j]);
            }
            *(v4f*)&Lr[c] = x;
        }
        mx = wredmax(mx);

        float sm = 0.f;
#pragma unroll 4
        for (int it = 0; it < 16; ++it) {
            const int c = it * 128 + ln * 4;
            v4f x = *(v4f*)&Lr[c];
#pragma unroll
            for (int j = 0; j < 4; ++j) { x[j] = __expf(x[j] - mx); sm += x[j]; }
            *(v4f*)&Lr[c] = x;
        }
        sm = wredsum(sm);
        const float inv = 1.0f / sm;

        float* Or = &attnOut[(size_t)(bh * SS + q) * SS];
#pragma unroll 4
        for (int it = 0; it < 16; ++it) {
            const int c = it * 128 + ln * 4;
            v4f x = *(v4f*)&Lr[c];
            x *= inv;
            *(v4f*)&Lr[c] = x;
            *(v4f*)&Or[c] = x;   // attn leaves HBM exactly once
        }
    }
    __syncthreads();

    // ---- Phase 3: ctx partials, K chunk [wv*256, wv*256+256) ----
    v8f acc[4] = {};
    const int kb0 = wv * 256;
#pragma unroll 2
    for (int kk = 0; kk < 256; kk += 32) {
        const int kb = kb0 + kk;
        union { v16bf v; v8bf p[2]; } a;
        a.p[0] = cvt8(*(const v8f*)&lgt[l15 * LROW + kb + lh * 16]);
        a.p[1] = cvt8(*(const v8f*)&lgt[l15 * LROW + kb + lh * 16 + 8]);
#pragma unroll
        for (int c = 0; c < 4; ++c) {
            const v16bf bf = *(const v16bf*)&Vb[(size_t)(c * 16 + l15) * SS + kb + lh * 16];
            acc[c] = wmma_bf16(a.v, bf, acc[c]);
        }
    }
#pragma unroll
    for (int c = 0; c < 4; ++c)
#pragma unroll
        for (int r = 0; r < 8; ++r)
            rbuf[wv * 1024 + (r + lh * 8) * 64 + c * 16 + l15] = acc[c][r];
    __syncthreads();

    // ---- tree-reduce partials, store merged-head bf16 ctx ----
    for (int o = tid; o < 1024; o += 256) {
        float s = 0.f;
#pragma unroll
        for (int w = 0; w < 8; ++w) s += rbuf[w * 1024 + o];
        const int mr = o >> 6, d = o & 63;
        ctxOut[(size_t)(b * SS + qBase + mr) * DD + h * HD + d] = (__bf16)s;
    }
}

// ---------------------------------------------------------------------------
extern "C" void kernel_launch(void* const* d_in, const int* in_sizes, int n_in,
                              void* d_out, int out_size, void* d_ws, size_t ws_size,
                              hipStream_t stream) {
    const float* v    = (const float*)d_in[0];
    const float* k    = (const float*)d_in[1];
    const float* q    = (const float*)d_in[2];
    const int*   mask = (const int*)  d_in[3];
    const float* Wq   = (const float*)d_in[4];
    const float* bq   = (const float*)d_in[5];
    const float* Wk   = (const float*)d_in[6];
    const float* bk   = (const float*)d_in[7];
    const float* Wv   = (const float*)d_in[8];
    const float* bv   = (const float*)d_in[9];
    const float* Wo   = (const float*)d_in[10];
    const float* bo   = (const float*)d_in[11];

    float* xOut    = (float*)d_out;
    float* attnOut = xOut + (size_t)BB * SS * DD;

    const size_t tElems = (size_t)BB * HH * SS * HD;   // 4,194,304
    __bf16* Qh  = (__bf16*)d_ws;
    __bf16* Kh  = Qh + tElems;
    __bf16* Vt  = Kh + tElems;
    __bf16* ctx = Vt + tElems;

    const dim3 blk(256);
    const dim3 gGemm(DD / 128, (BB * SS) / 128);       // (8, 32)

    gemm_wmma<float, 0><<<gGemm, blk, 0, stream>>>(q, Wq, bq, (void*)Qh);
    gemm_wmma<float, 0><<<gGemm, blk, 0, stream>>>(k, Wk, bk, (void*)Kh);
    gemm_wmma<float, 1><<<gGemm, blk, 0, stream>>>(v, Wv, bv, (void*)Vt);

    const size_t smemBytes = (size_t)16 * LROW * sizeof(float)   // logits slab
                           + (size_t)8 * 16 * 64 * sizeof(float); // ctx partials
    attn_fused<<<dim3(SS / 16, BB * HH), blk, smemBytes, stream>>>(
        Qh, Kh, Vt, mask, attnOut, ctx);

    gemm_wmma<__bf16, 2><<<gGemm, blk, 0, stream>>>(ctx, Wo, bo, (void*)xOut);
}
